// BuildLstmUnrollNet_66649302499803
// MI455X (gfx1250) — matrix operand order repaired and checked by reading, not compiled, not run on hardware
//
#include <hip/hip_runtime.h>
#include <hip/hip_bf16.h>

// ---------------------------------------------------------------------------
// 2-layer LSTM, T=11 (only 10 steps reach the output), B=8192, R=425, IN=20.
// Batch-parallel: BM=32 rows per 512-thread workgroup (16 waves, 4/SIMD).
// Each wave owns 7 consecutive gate tiles; gates via v_wmma_f32_16x16x32_f16.
// B fragments are streamed global->LDS with global_load_async_to_lds_b128
// (2-slot per-wave ring, ASYNCcnt-paced) so in-flight bytes are not capped by
// VGPRs; the ring aliases the gates buffer (dead during the kt loop).
// h lives in LDS in WMMA-A fragment order; c state in VGPRs.
// ---------------------------------------------------------------------------

#define T_STEPS 10
#define NTILES  112      // gate tiles of 16 (1700 -> 107 real, padded to 112)
#define KT_R    14       // K tiles of 32 for R=425 (padded to 448)
#define GATES   1700
#define RDIM    425
#define INDIM   20
#define BM      32
#define BATCH   8192
#define OUTW    (11 * RDIM)     // 4675
#define THREADS 512
#define RUN     7               // gate tiles per wave
#define CELEM   27              // ceil(BM*RDIM / THREADS)
#define BSUMW   (NTILES * 16)   // 1792
// per-wave B ring: 2 slots x 7 frags x 1024B = 14336 B;  16 waves = 229376 B
#define SLOT_B  7168
#define RING_B  14336
#define DYN_B   (16 * RING_B)

typedef _Float16 v16h __attribute__((ext_vector_type(16)));
typedef float    v8f  __attribute__((ext_vector_type(8)));

#define WMMA_F16(A, B, C) \
    __builtin_amdgcn_wmma_f32_16x16x32_f16(false, (A), false, (B), (short)0, (C), false, false)

__device__ __forceinline__ float fsigm(float x) {
    return __builtin_amdgcn_rcpf(1.0f + __builtin_amdgcn_exp2f(-1.44269504f * x));
}
__device__ __forceinline__ float ftanh_(float x) {
    return 2.0f * __builtin_amdgcn_rcpf(1.0f + __builtin_amdgcn_exp2f(-2.88539008f * x)) - 1.0f;
}

__device__ __forceinline__ v16h loadv16(const _Float16* p) { return *(const v16h*)p; }

// A fragment: h pre-swizzled [sub][kt][lane][i] -> contiguous 32B per lane
__device__ __forceinline__ v16h load_a(const _Float16* sb, int sub, int kt, int lane) {
    return *(const v16h*)(sb + (size_t)(((sub * KT_R + kt) * 32) + lane) * 16);
}

// Issue one 7KB B batch (7 contiguous fragments) as 14 async b128 copies.
__device__ __forceinline__ void async_batch(const _Float16* g, unsigned ldsbyte, int lane) {
    unsigned long long ga = (unsigned long long)(size_t)g + (unsigned)(lane * 16);
    unsigned la = ldsbyte + (unsigned)(lane * 16);
    asm volatile("s_wait_xcnt 0" ::: "memory");   // addr VGPR reuse vs in-flight
#pragma unroll
    for (int c = 0; c < 14; ++c)
        asm volatile("global_load_async_to_lds_b128 %0, %1, off offset:%2"
                     :: "v"(la), "v"(ga), "i"(c * 512) : "memory");
}

// Global base of batch bb for this wave (weights laid out [t][kt][nt][512]).
template <bool L1>
__device__ __forceinline__ const _Float16* batch_base(
    int bb, int t, int ntbase,
    const _Float16* Wx, const _Float16* W1, const _Float16* W2)
{
    if (!L1) {
        if (bb == 0) return Wx + ((size_t)t * NTILES + ntbase) * 512;
        return W1 + (((size_t)t * KT_R + (bb - 1)) * NTILES + ntbase) * 512;
    } else {
        if (bb < KT_R) return W1 + (((size_t)t * KT_R + bb) * NTILES + ntbase) * 512;
        return W2 + (((size_t)t * KT_R + (bb - KT_R)) * NTILES + ntbase) * 512;
    }
}

// D 16x16 f32 layout: lane l: N=l&15; elem v: M=v+((l&16)?8:0).  sg = [32][1700] f16.
__device__ __forceinline__ void store_g(_Float16* sg, v8f aS0, v8f aS1, int nt, int lane) {
    int n = nt * 16 + (lane & 15);
    if (n < GATES) {
        int mb = (lane & 16) ? 8 : 0;
#pragma unroll
        for (int v = 0; v < 8; ++v) {
            sg[(size_t)(mb + v) * GATES + n]      = (_Float16)aS0[v];
            sg[(size_t)(16 + mb + v) * GATES + n] = (_Float16)aS1[v];
        }
    }
}

// One layer's gate GEMM with async-LDS B ring.  bst doubles as gates buffer.
template <bool L1>
__device__ __forceinline__ void gates_gemm(
    int t, int wave, int lane,
    const float* __restrict__ bsum,
    const _Float16* Wx, const _Float16* W1, const _Float16* W2,
    const _Float16* __restrict__ sx,
    const _Float16* __restrict__ shA,   // sh0
    const _Float16* __restrict__ shB,   // sh1 (layer 1 only)
    _Float16* bst)
{
    constexpr int NB = L1 ? (2 * KT_R) : (KT_R + 1);
    const int ntbase = RUN * wave;                  // wave-uniform
    const int n0 = ntbase * 16 + (lane & 15);
    const unsigned ringbyte = (unsigned)(size_t)bst + (unsigned)(wave * RING_B);
    const _Float16* ring = bst + (size_t)wave * (RING_B / 2);

    // prologue: start batches 0 and 1
    async_batch(batch_base<L1>(0, t, ntbase, Wx, W1, W2), ringbyte, lane);
    async_batch(batch_base<L1>(1, t, ntbase, Wx, W1, W2), ringbyte + SLOT_B, lane);

    v8f acc[RUN][2];
#pragma unroll
    for (int q = 0; q < RUN; ++q) {
        float bv = bsum[(size_t)t * BSUMW + n0 + q * 16];
#pragma unroll
        for (int v = 0; v < 8; ++v) { acc[q][0][v] = bv; acc[q][1][v] = bv; }
    }

#pragma unroll
    for (int bb = 0; bb < NB; ++bb) {
        // in-order async completion: <=14 outstanding => batch bb has landed
        if (bb + 1 < NB) asm volatile("s_wait_asynccnt 14" ::: "memory");
        else             asm volatile("s_wait_asynccnt 0"  ::: "memory");

        v16h a0, a1;
        if (!L1) {
            if (bb == 0) {
                a0 = loadv16(sx + (size_t)lane * 16);
                a1 = loadv16(sx + (size_t)(32 + lane) * 16);
            } else {
                a0 = load_a(shA, 0, bb - 1, lane);
                a1 = load_a(shA, 1, bb - 1, lane);
            }
        } else {
            if (bb < KT_R) { a0 = load_a(shA, 0, bb, lane);        a1 = load_a(shA, 1, bb, lane); }
            else           { a0 = load_a(shB, 0, bb - KT_R, lane); a1 = load_a(shB, 1, bb - KT_R, lane); }
        }
        const _Float16* bs = ring + (size_t)(bb & 1) * (SLOT_B / 2) + (size_t)lane * 16;
#pragma unroll
        for (int q = 0; q < RUN; ++q) {
            v16h b = loadv16(bs + (size_t)q * 512);
            acc[q][0] = WMMA_F16(a0, b, acc[q][0]);
            acc[q][1] = WMMA_F16(a1, b, acc[q][1]);
        }
        if (bb + 2 < NB)    // refill the slot just consumed
            async_batch(batch_base<L1>(bb + 2, t, ntbase, Wx, W1, W2),
                        ringbyte + (unsigned)((bb & 1) * SLOT_B), lane);
    }

    __syncthreads();   // all waves done with the B ring before gates alias it
#pragma unroll
    for (int q = 0; q < RUN; ++q)
        store_g(bst, acc[q][0], acc[q][1], ntbase + q, lane);
}

// LSTM cell elementwise; c state in registers.  h written back into the
// swizzled A-fragment layout; layer-1 also writes the (coalesced) output.
template <bool L1>
__device__ __forceinline__ void cell_update(
    const _Float16* __restrict__ sg, _Float16* __restrict__ shh,
    float* cr, float* __restrict__ out, int b0, int trow, int tid)
{
#pragma unroll
    for (int kk = 0; kk < CELEM; ++kk) {
        int e = tid + THREADS * kk;
        if (e < BM * RDIM) {
            int b = e / RDIM, j = e - b * RDIM;
            const _Float16* gr = sg + (size_t)b * GATES + j;
            float n1 = (float)gr[0];
            float n2 = (float)gr[RDIM];
            float n3 = (float)gr[2 * RDIM];
            float n4 = (float)gr[3 * RDIM];
            float ig = fsigm(n1), fg = fsigm(n2), og = fsigm(n3), gg = ftanh_(n4);
            float c = fg * cr[kk] + ig * gg;
            cr[kk] = c;
            float h = og * ftanh_(c);
            int sub = b >> 4, m = b & 15, kt = j >> 5, k = j & 31;
            int off = (((sub * KT_R + kt) * 32) + (m | ((k & 8) << 1))) * 16
                      + ((k & 7) | ((k & 16) >> 1));
            shh[off] = (_Float16)h;
            if (L1) out[(size_t)(b0 + b) * OUTW + trow + j] = h;
        }
    }
}

__launch_bounds__(THREADS)
__global__ void lstm_kernel(
    const float* __restrict__ x, const float* __restrict__ init,
    const _Float16* __restrict__ Wi0s, const _Float16* __restrict__ Wh0s,
    const _Float16* __restrict__ Wi1s, const _Float16* __restrict__ Wh1s,
    const float* __restrict__ bsum0, const float* __restrict__ bsum1,
    float* __restrict__ out)
{
    __shared__ __align__(32) _Float16 sh0[2 * KT_R * 32 * 16];   // 28672 B
    __shared__ __align__(32) _Float16 sh1[2 * KT_R * 32 * 16];   // 28672 B
    __shared__ __align__(32) _Float16 sx[2 * 32 * 16];           //  2048 B
    extern __shared__ __align__(64) _Float16 bst[];              // 229376 B ring / gates alias

    const int tid  = threadIdx.x;
    const int lane = tid & 31;
    const int wave = tid >> 5;
    const int b0   = blockIdx.x * BM;

    for (int i = tid; i < 2 * KT_R * 32 * 16; i += THREADS) {
        sh0[i] = (_Float16)0.0f;  sh1[i] = (_Float16)0.0f;
    }
    for (int i = tid; i < 2 * 32 * 16; i += THREADS) {
        int s = i >> 9, lane2 = (i >> 4) & 31, ii = i & 15;
        int m = lane2 & 15;
        int k = (ii & 7) | ((lane2 & 16) >> 1) | ((ii & 8) << 1);
        float v = 0.0f;
        if (k < INDIM) v = x[(size_t)(b0 + s * 16 + m) * INDIM + k];
        sx[i] = (_Float16)v;
    }
    __syncthreads();

    float c0r[CELEM], c1r[CELEM];
#pragma unroll
    for (int kk = 0; kk < CELEM; ++kk) {
        c0r[kk] = 0.0f;  c1r[kk] = 0.0f;
        int e = tid + THREADS * kk;
        if (e < BM * RDIM) {
            int b = e / RDIM, j = e - b * RDIM;
            const float* ib = init + (size_t)(b0 + b) * (4 * RDIM);
            float h0v = ib[j], c0v = ib[RDIM + j], h1v = ib[2 * RDIM + j], c1v = ib[3 * RDIM + j];
            c0r[kk] = c0v;  c1r[kk] = c1v;
            int sub = b >> 4, m = b & 15, kt = j >> 5, k = j & 31;
            int off = (((sub * KT_R + kt) * 32) + (m | ((k & 8) << 1))) * 16
                      + ((k & 7) | ((k & 16) >> 1));
            sh0[off] = (_Float16)h0v;
            sh1[off] = (_Float16)h1v;
            out[(size_t)(b0 + b) * OUTW + j] = h1v;
        }
    }

#pragma unroll 1
    for (int t = 0; t < T_STEPS; ++t) {
        __syncthreads();
        gates_gemm<false>(t, wave, lane, bsum0, Wi0s, Wh0s, nullptr, sx, sh0, nullptr, bst);
        __syncthreads();
        cell_update<false>(bst, sh0, c0r, out, b0, 0, tid);
        __syncthreads();
        gates_gemm<true>(t, wave, lane, bsum1, nullptr, Wi1s, Wh1s, sx, sh0, sh1, bst);
        __syncthreads();
        cell_update<true>(bst, sh1, c1r, out, b0, (t + 1) * RDIM, tid);
    }
}

// ---------------------------------------------------------------------------
// Prep: fp32 weights -> f16 WMMA-B fragments, layout [t][kt][nt][lane][i]
// so each wave's per-kt batch (7 tiles) is one contiguous 7KB block.
// ---------------------------------------------------------------------------
__global__ void swizzle_weights(const float* __restrict__ W, _Float16* __restrict__ dst,
                                int K, int KTT, long total)
{
    long idx = (long)blockIdx.x * 256 + threadIdx.x;
    if (idx >= total) return;
    int  i    = (int)(idx & 15);
    int  lane = (int)((idx >> 4) & 31);
    long rest = idx >> 9;
    int  nt   = (int)(rest % NTILES);  rest /= NTILES;
    int  kt   = (int)(rest % KTT);
    int  t    = (int)(rest / KTT);
    int  n    = nt * 16 + (lane & 15);
    int  k    = kt * 32 + ((lane & 16) ? 8 : 0) + (i & 7) + ((i & 8) << 1);
    float v = 0.0f;
    if (n < GATES && k < K) v = W[((size_t)t * GATES + n) * K + k];
    dst[idx] = (_Float16)v;
}

__global__ void prep_bias(const float* __restrict__ bi0, const float* __restrict__ bh0,
                          const float* __restrict__ bi1, const float* __restrict__ bh1,
                          float* __restrict__ bsum0, float* __restrict__ bsum1)
{
    int idx = blockIdx.x * 256 + threadIdx.x;
    if (idx >= T_STEPS * BSUMW) return;
    int t = idx / BSUMW, n = idx - t * BSUMW;
    float v0 = 0.0f, v1 = 0.0f;
    if (n < GATES) {
        v0 = bi0[(size_t)t * GATES + n] + bh0[(size_t)t * GATES + n];
        v1 = bi1[(size_t)t * GATES + n] + bh1[(size_t)t * GATES + n];
    }
    bsum0[idx] = v0;  bsum1[idx] = v1;
}

extern "C" void kernel_launch(void* const* d_in, const int* in_sizes, int n_in,
                              void* d_out, int out_size, void* d_ws, size_t ws_size,
                              hipStream_t stream)
{
    (void)in_sizes; (void)n_in; (void)out_size; (void)ws_size;
    const float* x    = (const float*)d_in[0];
    const float* init = (const float*)d_in[1];
    const float* Wi0  = (const float*)d_in[2];
    const float* bi0  = (const float*)d_in[3];
    const float* Wh0  = (const float*)d_in[4];
    const float* bh0  = (const float*)d_in[5];
    const float* Wi1  = (const float*)d_in[6];
    const float* bi1  = (const float*)d_in[7];
    const float* Wh1  = (const float*)d_in[8];
    const float* bh1  = (const float*)d_in[9];
    float* out = (float*)d_out;

    char* ws = (char*)d_ws;
    const size_t szWi0 = (size_t)T_STEPS * NTILES * 1    * 512 * 2;
    const size_t szWh  = (size_t)T_STEPS * NTILES * KT_R * 512 * 2;
    const size_t szB   = (size_t)T_STEPS * BSUMW * 4;
    _Float16* Wi0s = (_Float16*)ws;  ws += szWi0;
    _Float16* Wh0s = (_Float16*)ws;  ws += szWh;
    _Float16* Wi1s = (_Float16*)ws;  ws += szWh;
    _Float16* Wh1s = (_Float16*)ws;  ws += szWh;
    float* bsum0 = (float*)ws;       ws += szB;
    float* bsum1 = (float*)ws;

    const long totWi0 = (long)T_STEPS * NTILES * 1    * 512;
    const long totWh  = (long)T_STEPS * NTILES * KT_R * 512;
    swizzle_weights<<<(unsigned)((totWi0 + 255) / 256), 256, 0, stream>>>(Wi0, Wi0s, INDIM, 1, totWi0);
    swizzle_weights<<<(unsigned)((totWh  + 255) / 256), 256, 0, stream>>>(Wh0, Wh0s, RDIM, KT_R, totWh);
    swizzle_weights<<<(unsigned)((totWh  + 255) / 256), 256, 0, stream>>>(Wi1, Wi1s, RDIM, KT_R, totWh);
    swizzle_weights<<<(unsigned)((totWh  + 255) / 256), 256, 0, stream>>>(Wh1, Wh1s, RDIM, KT_R, totWh);
    const int totB = T_STEPS * BSUMW;
    prep_bias<<<(totB + 255) / 256, 256, 0, stream>>>(bi0, bh0, bi1, bh1, bsum0, bsum1);

    (void)hipFuncSetAttribute((const void*)lstm_kernel,
                              hipFuncAttributeMaxDynamicSharedMemorySize, DYN_B);
    lstm_kernel<<<BATCH / BM, THREADS, DYN_B, stream>>>(
        x, init, Wi0s, Wh0s, Wi1s, Wh1s, bsum0, bsum1, out);
}
